// ScaledDotProductAttention_73254962200802
// MI455X (gfx1250) — compile-verified, matrix operand order checked by reading, not run on hardware
//
#include <hip/hip_runtime.h>
#include <cstdint>

// ---------------------------------------------------------------------------
// Scaled dot-product attention for MI455X (gfx1250), wave32 + WMMA f16 +
// async global->LDS copies (GLOBAL_LOAD_ASYNC_TO_LDS_B128, ASYNCcnt).
// Outputs (concatenated in d_out): context [B,H,S,DK] fp32, attn [B,H,S,S] fp32.
// One wave (32 threads) per workgroup handles a 16-row Q tile of one (b,h).
// ---------------------------------------------------------------------------

#define B_DIM  2
#define H_DIM  16
#define S_LEN  2048
#define DK_DIM 64

typedef __attribute__((ext_vector_type(16))) _Float16 v16h;
typedef __attribute__((ext_vector_type(2)))  _Float16 h2;
typedef __attribute__((ext_vector_type(2)))  float    f2;
typedef __attribute__((ext_vector_type(8)))  float    v8f;

#define WMMA_F16(A, B, C) \
  __builtin_amdgcn_wmma_f32_16x16x32_f16(false, (A), false, (B), (short)0, (C), false, false)

// s_wait_asynccnt <= n  (async loads complete in order)
#define ASYNC_WAIT(n) asm volatile("s_wait_asynccnt " #n ::: "memory")

// LDS byte offset of a __shared__ object (addrspacecast flat -> local).
__device__ __forceinline__ unsigned lds_off_of(const void* p) {
  return (unsigned)(uintptr_t)(__attribute__((address_space(3))) const void*)p;
}

// Issue NBYTES/512 async copies: each lane moves 16 contiguous bytes per op.
// Source must be NBYTES contiguous in global memory.
template <int NBYTES>
__device__ __forceinline__ void async_load_tile(unsigned lds_off, const float* gsrc, int lane) {
  unsigned long long gbase = (unsigned long long)(uintptr_t)gsrc;  // uniform -> SGPR pair
#pragma unroll
  for (int i = 0; i < NBYTES; i += 512) {
    unsigned voff  = (unsigned)(lane * 16 + i);
    unsigned laddr = lds_off + voff;
    asm volatile("global_load_async_to_lds_b128 %0, %1, %2"
                 :: "v"(laddr), "v"(voff), "s"(gbase)
                 : "memory");
  }
}

// Murmur-style 64->32 mix for a deterministic dropout keep-mask (p = 0.5).
__device__ __forceinline__ unsigned int mix64(unsigned long long x) {
  x ^= x >> 33; x *= 0xff51afd7ed558ccdULL;
  x ^= x >> 33; x *= 0xc4ceb9fe1a85ec53ULL;
  x ^= x >> 33;
  return (unsigned int)x;
}

// 16-bit WMMA operand layout (wave32): lane holds row (M or N, 0..15), VGPR j
// holds the f16 pair at K = ((j>>2)<<4) + ((j&3)<<1) + (half<<3) + kbase.
__device__ __forceinline__ int kidx(int j, int half) {
  return ((j >> 2) << 4) + ((j & 3) << 1) + (half << 3);
}

// Operand from an f16 LDS tile (row-major, given stride): packed b32 reads.
__device__ __forceinline__ v16h ld_op_f16(const _Float16* __restrict__ s,
                                          int row, int kbase, int half, int stride) {
  v16h o;
  const h2* p2 = (const h2*)s;
#pragma unroll
  for (int j = 0; j < 8; ++j) {
    int k = kbase + kidx(j, half);
    h2 v = p2[(row * stride + k) >> 1];
    o[2 * j] = v.x; o[2 * j + 1] = v.y;
  }
  return o;
}

// Operand from an f32 LDS tile (row-major, stride 64): b64 read + cvt to f16.
__device__ __forceinline__ v16h ld_op_f32(const float* __restrict__ s,
                                          int row, int kbase, int half) {
  v16h o;
  const f2* p2 = (const f2*)s;
#pragma unroll
  for (int j = 0; j < 8; ++j) {
    int k = kbase + kidx(j, half);
    f2 v = p2[(row * 64 + k) >> 1];
    o[2 * j] = (_Float16)v.x; o[2 * j + 1] = (_Float16)v.y;
  }
  return o;
}

// B operand gathering a column of a row-major f32 V tile (K dim = rows).
__device__ __forceinline__ v16h ld_opB_vcol(const float* __restrict__ s,
                                            int col, int half) {
  v16h o;
#pragma unroll
  for (int j = 0; j < 8; ++j) {
    int k = kidx(j, half);
    o[2 * j]     = (_Float16)s[k * 64 + col];
    o[2 * j + 1] = (_Float16)s[(k + 1) * 64 + col];
  }
  return o;
}

__device__ __forceinline__ float rmax16(float v) {
#pragma unroll
  for (int o = 8; o > 0; o >>= 1) v = fmaxf(v, __shfl_xor(v, o, 32));
  return v;
}
__device__ __forceinline__ float rsum16(float v) {
#pragma unroll
  for (int o = 8; o > 0; o >>= 1) v += __shfl_xor(v, o, 32);
  return v;
}

__global__ void __launch_bounds__(32)
sdpa_wmma_kernel(const float* __restrict__ Q, const float* __restrict__ K,
                 const float* __restrict__ V, const float* __restrict__ mask,
                 float* __restrict__ ctx, float* __restrict__ attn) {
  const int lane = threadIdx.x;        // 0..31
  const int half = lane >> 4;          // 0 | 1
  const int nl   = lane & 15;          // 0..15
  const int qt = blockIdx.x;           // q tile (16 rows)
  const int h  = blockIdx.y;
  const int b  = blockIdx.z;
  const int q0 = qt * 16;

  const size_t bh = (size_t)b * H_DIM + h;
  const float* Qp = Q + (bh * S_LEN + q0) * DK_DIM;
  const float* Kp = K + bh * S_LEN * DK_DIM;
  const float* Vp = V + bh * S_LEN * DK_DIM;
  const float* Mp = mask + (size_t)b * S_LEN * S_LEN;   // [B,1,S,S]

  __shared__ __align__(16) _Float16 sQ[16 * 64];     // 2 KB, f16
  __shared__ __align__(16) float    sKf[2][16 * 64]; // 2 x 4 KB, raw f32 (async)
  __shared__ __align__(16) float    sVf[32 * 64];    // 8 KB, raw f32 (async)
  __shared__ __align__(16) _Float16 sP[16 * 32];     // 1 KB, P transpose

  const unsigned kofs0 = lds_off_of(&sKf[0][0]);
  const unsigned kofs1 = lds_off_of(&sKf[1][0]);
  const unsigned vofs  = lds_off_of(&sVf[0]);

  // ---- stage Q tile once (fp32 -> f16) ----
  for (int i = lane; i < 16 * 64; i += 32) {
    int r = i >> 6, c = i & 63;
    sQ[i] = (_Float16)Qp[(size_t)r * DK_DIM + c];
  }
  __builtin_amdgcn_wave_barrier();
  const v16h aQ0 = ld_op_f16(sQ, nl, 0, half, 64);   // k = 0..31
  const v16h aQ1 = ld_op_f16(sQ, nl, 32, half, 64);  // k = 32..63

  const float kscale = 0.125f;  // 1/sqrt(64)

  // Score tile from an LDS-resident f32 K tile; D layout:
  // row = q0 + r + 8*half, col = n0 + nl.
  auto score_tile = [&](const float* kbuf, int n0, float s[8]) {
    v16h b0 = ld_op_f32(kbuf, nl, 0, half);
    v16h b1 = ld_op_f32(kbuf, nl, 32, half);
    v8f a = {};
    a = WMMA_F16(aQ0, b0, a);
    a = WMMA_F16(aQ1, b1, a);
#pragma unroll
    for (int r = 0; r < 8; ++r) {
      int row = q0 + r + 8 * half;
      float mv = Mp[(size_t)row * S_LEN + n0 + nl];
      s[r] = a[r] * kscale + (mv - 1.0f) * 1.0e9f;
    }
  };

  // ---------------- pass 1: online row-max m and row-sumexp l ----------------
  float m[8], l[8];
#pragma unroll
  for (int r = 0; r < 8; ++r) { m[r] = -3.0e38f; l[r] = 0.0f; }

  async_load_tile<4096>(kofs0, Kp, lane);  // prime tile 0
  for (int nt = 0; nt < S_LEN / 16; ++nt) {
    if (nt + 1 < S_LEN / 16) {   // prefetch next K tile into the other buffer
      async_load_tile<4096>(((nt + 1) & 1) ? kofs1 : kofs0,
                            Kp + (size_t)(nt + 1) * 16 * DK_DIM, lane);
      ASYNC_WAIT(0x8);           // current tile's 8 ops done (in-order)
    } else {
      ASYNC_WAIT(0x0);
    }
    float s[8];
    score_tile(sKf[nt & 1], nt * 16, s);
#pragma unroll
    for (int r = 0; r < 8; ++r) {
      float tmax = rmax16(s[r]);
      float mnew = fmaxf(m[r], tmax);
      float e    = __expf(s[r] - mnew);
      float tsum = rsum16(e);
      l[r] = l[r] * __expf(m[r] - mnew) + tsum;
      m[r] = mnew;
    }
  }

  float rl[8];
#pragma unroll
  for (int r = 0; r < 8; ++r) rl[r] = (l[r] > 0.0f) ? (1.0f / l[r]) : 0.0f;

  // -------- pass 2: attn write, dropout, and context = P_drop @ V ----------
  v8f acc[4];
#pragma unroll
  for (int dt = 0; dt < 4; ++dt) { v8f z = {}; acc[dt] = z; }

  for (int chunk = 0; chunk < S_LEN / 32; ++chunk) {
    const int nc0 = chunk * 32;
    // all chunk data up front: V 8KB + 2 K tiles, one wait
    async_load_tile<8192>(vofs,  Vp + (size_t)nc0 * DK_DIM, lane);
    async_load_tile<4096>(kofs0, Kp + (size_t)nc0 * DK_DIM, lane);
    async_load_tile<4096>(kofs1, Kp + (size_t)(nc0 + 16) * DK_DIM, lane);
    ASYNC_WAIT(0x0);

#pragma unroll
    for (int t = 0; t < 2; ++t) {
      const int n0 = nc0 + t * 16;
      float s[8];
      score_tile(sKf[t], n0, s);
#pragma unroll
      for (int r = 0; r < 8; ++r) {
        int row  = q0 + r + 8 * half;
        float p  = __expf(s[r] - m[r]) * rl[r];
        size_t fidx = (bh * S_LEN + (size_t)row) * S_LEN + (size_t)(n0 + nl);
        attn[fidx] = p;                       // pre-dropout softmax (fp32 out)
        unsigned int hv = mix64(fidx ^ 0x9E3779B97F4A7C15ULL);
        float pd = (hv & 1u) ? (p * 2.0f) : 0.0f;  // p_drop = 0.5, x1/(1-p)
        sP[(r + 8 * half) * 32 + t * 16 + nl] = (_Float16)pd;
      }
    }
    __builtin_amdgcn_wave_barrier();
    v16h aP = ld_op_f16(sP, nl, 0, half, 32);       // A: 16 x 32 (K = n-chunk)
#pragma unroll
    for (int dt = 0; dt < 4; ++dt) {
      v16h bV = ld_opB_vcol(sVf, dt * 16 + nl, half);  // B: 32 x 16 d-cols
      acc[dt] = WMMA_F16(aP, bV, acc[dt]);
    }
    __builtin_amdgcn_wave_barrier();
  }

  // ---- store context (fp32, D layout -> global) ----
#pragma unroll
  for (int dt = 0; dt < 4; ++dt) {
#pragma unroll
    for (int r = 0; r < 8; ++r) {
      int row = q0 + r + 8 * half;
      ctx[(bh * S_LEN + (size_t)row) * DK_DIM + dt * 16 + nl] = acc[dt][r];
    }
  }
}

extern "C" void kernel_launch(void* const* d_in, const int* in_sizes, int n_in,
                              void* d_out, int out_size, void* d_ws, size_t ws_size,
                              hipStream_t stream) {
  const float* Q    = (const float*)d_in[0];
  const float* K    = (const float*)d_in[1];
  const float* V    = (const float*)d_in[2];
  const float* mask = (const float*)d_in[3];

  float* ctx  = (float*)d_out;
  float* attn = ctx + (size_t)B_DIM * H_DIM * S_LEN * DK_DIM;  // outputs concatenated

  dim3 grid(S_LEN / 16, H_DIM, B_DIM);  // 128 x 16 x 2 = 4096 workgroups
  dim3 block(32);                       // one wave32 per workgroup
  hipLaunchKernelGGL(sdpa_wmma_kernel, grid, block, 0, stream,
                     Q, K, V, mask, ctx, attn);

  (void)in_sizes; (void)n_in; (void)out_size; (void)d_ws; (void)ws_size;
}
